// FGNNBasedFormulaEmbeddingLayer_75831942578506
// MI455X (gfx1250) — matrix operand order repaired and checked by reading, not compiled, not run on hardware
//
#include <hip/hip_runtime.h>
#include <hip/hip_bf16.h>

#define N_ATOMS 50000
#define D 256
#define Fdim 4
#define Gdim 50000
#define Adim 3
#define IN_DIM 512
#define KSTEPS (IN_DIM / 32)
#define MTILES ((Gdim * Adim) / 16)   // 9375 exact

typedef __attribute__((ext_vector_type(16))) __bf16 v16bf;
typedef __attribute__((ext_vector_type(8)))  float  v8f;

union FragBF { uint4 q[2]; v16bf v; };

__device__ __forceinline__ unsigned short f2bf(float f) {
    unsigned int u = __float_as_uint(f);
    u += 0x7FFFu + ((u >> 16) & 1u);          // round-to-nearest-even
    return (unsigned short)(u >> 16);
}

// B frag (32x16 bf16): elems 0-15 -> K = kb + lh*16 + e (lh folded into p)
__device__ __forceinline__ void loadB(FragBF& b, const unsigned short* p, int kb) {
    b.q[0] = *(const uint4*)(p + kb);
    b.q[1] = *(const uint4*)(p + kb + 8);
}
// A frag (16x32 bf16): elems 0-15 -> K = kb + lh*16 + e
__device__ __forceinline__ void loadA(FragBF& a, const unsigned short* arow, int kb, int lh) {
    a.q[0] = *(const uint4*)(arow + kb + lh * 16);
    a.q[1] = *(const uint4*)(arow + kb + lh * 16 + 8);
}

// ---------------------------------------------------------------------------
// Convert weights to bf16, transposed to [f][n][k] so B-fragments are
// K-contiguous 16-byte chunks per lane.
// sel 0: W_M layer0, sel 1: W_U layer0, sel 2: W_U layer1
// ---------------------------------------------------------------------------
__global__ void prep_weights(const float* __restrict__ W_M,
                             const float* __restrict__ W_U,
                             unsigned short* __restrict__ WT_M0,
                             unsigned short* __restrict__ WT_U0,
                             unsigned short* __restrict__ WT_U1)
{
    const int S = IN_DIM * D;                 // 131072 per formula
    int gid = blockIdx.x * 256 + threadIdx.x;
    if (gid >= 3 * Fdim * S) return;
    int sel = gid / (Fdim * S);
    int rem = gid - sel * (Fdim * S);
    int f = rem / S;
    int e = rem - f * S;                      // e = n*512 + k
    int n = e >> 9;
    int k = e & 511;
    const float* src = (sel == 0) ? W_M : (sel == 1 ? W_U : (W_U + (size_t)Fdim * S));
    float v = src[(size_t)f * S + (size_t)k * D + n];
    unsigned short* dst = (sel == 0) ? WT_M0 : (sel == 1 ? WT_U0 : WT_U1);
    dst[rem] = f2bf(v);
}

__global__ void zero4(float4* __restrict__ p, int n4)
{
    int i = blockIdx.x * blockDim.x + threadIdx.x;
    int stride = gridDim.x * blockDim.x;
    float4 z = {0.f, 0.f, 0.f, 0.f};
    for (; i < n4; i += stride) p[i] = z;
}

// ---------------------------------------------------------------------------
// One layer: gather+concat -> bf16 A-tile in LDS -> WMMA GEMM against
// transposed bf16 weights (software-pipelined) -> bias+ReLU -> atomic-max.
// HAS_M: also compute message transform (clique-max path), layer 0 only.
// Grid: (MTILES, Fdim), block 256 = 8 waves; wave w owns columns [32w,32w+32).
// ---------------------------------------------------------------------------
template<bool HAS_M>
__global__ __launch_bounds__(256, 2)
void fgnn_layer(const float* __restrict__ atomSrc,        // [N_ATOMS, D]
                const float* __restrict__ gclSrc,         // [F, G, D]
                const int*   __restrict__ gidx,           // [F, G, 3]
                const unsigned short* __restrict__ WT_M,  // [F, D, IN_DIM] bf16
                const float* __restrict__ bM,             // [F, D]
                const unsigned short* __restrict__ WT_U,  // [F, D, IN_DIM] bf16
                const float* __restrict__ bU,             // [F, D]
                float* __restrict__ outGcl,               // [F, G, D] atomic-max
                float* __restrict__ outAtom)              // [N_ATOMS, D] atomic-max
{
    __shared__ __align__(16) unsigned short Atile[16][IN_DIM + 8]; // stride 1040B: conflict-free
    __shared__ int rowIdx[16];
    __shared__ int rowG[16];

    const int f = blockIdx.y;
    const int tid = threadIdx.x;
    const int lane = tid & 31;
    const int wave = tid >> 5;
    const int rowBase = blockIdx.x * 16;

    if (tid < 16) {
        int M = rowBase + tid;
        int g = M / 3;
        int a = M - 3 * g;
        rowG[tid] = g;
        rowIdx[tid] = gidx[((size_t)f * Gdim + g) * 3 + a];
    }
    __syncthreads();

    // Gather 16 rows x 512 cols (atom row | clique state), fp32 -> bf16 -> LDS.
    {
        int row = tid >> 4;                   // 16 threads per row
        int cb = (tid & 15) << 5;             // 32 columns per thread
        const float* src = (cb < D)
            ? (atomSrc + (size_t)rowIdx[row] * D + cb)
            : (gclSrc + ((size_t)f * Gdim + rowG[row]) * D + (cb - D));
        ushort4* dst = (ushort4*)&Atile[row][cb];
        #pragma unroll
        for (int j = 0; j < 8; ++j) {
            float4 v = ((const float4*)src)[j];
            ushort4 h = { f2bf(v.x), f2bf(v.y), f2bf(v.z), f2bf(v.w) };
            dst[j] = h;
        }
    }
    __syncthreads();

    const int lh = lane >> 4;                 // lane half (0/1)
    const int ln = lane & 15;
    const int col0 = wave * 32;

    const unsigned short* arow = &Atile[ln][0];
    const unsigned short* pU0 = WT_U + ((size_t)f * D + col0 + ln) * IN_DIM + lh * 16;
    const unsigned short* pU1 = pU0 + 16 * IN_DIM;
    const unsigned short* pM0 = HAS_M ? (WT_M + ((size_t)f * D + col0 + ln) * IN_DIM + lh * 16)
                                      : (const unsigned short*)nullptr;
    const unsigned short* pM1 = HAS_M ? (pM0 + 16 * IN_DIM) : (const unsigned short*)nullptr;

    v8f accU0 = {}, accU1 = {}, accM0 = {}, accM1 = {};

    // Software pipeline: fragments for step ks+1 are in flight while the
    // WMMAs for step ks execute, so each WMMA waits only on loads issued a
    // full iteration earlier (loadcnt retires in order).
    FragBF a, b0, b1, c0, c1;
    loadA(a, arow, 0, lh);
    loadB(b0, pU0, 0);
    loadB(b1, pU1, 0);
    if (HAS_M) { loadB(c0, pM0, 0); loadB(c1, pM1, 0); }

    #pragma unroll 4
    for (int ks = 0; ks < KSTEPS; ++ks) {
        // branch-free tail: last iteration redundantly re-loads itself
        int kn = (ks + 1 < KSTEPS ? ks + 1 : ks) * 32;
        FragBF an, b0n, b1n, c0n, c1n;
        loadA(an, arow, kn, lh);
        loadB(b0n, pU0, kn);
        loadB(b1n, pU1, kn);
        if (HAS_M) { loadB(c0n, pM0, kn); loadB(c1n, pM1, kn); }

        accU0 = __builtin_amdgcn_wmma_f32_16x16x32_bf16(false, a.v, false, b0.v, (short)0, accU0, false, false);
        accU1 = __builtin_amdgcn_wmma_f32_16x16x32_bf16(false, a.v, false, b1.v, (short)0, accU1, false, false);
        if (HAS_M) {
            accM0 = __builtin_amdgcn_wmma_f32_16x16x32_bf16(false, a.v, false, c0.v, (short)0, accM0, false, false);
            accM1 = __builtin_amdgcn_wmma_f32_16x16x32_bf16(false, a.v, false, c1.v, (short)0, accM1, false, false);
        }

        a = an; b0 = b0n; b1 = b1n;
        if (HAS_M) { c0 = c0n; c1 = c1n; }
    }

    // Epilogue: bias + ReLU, atomic max scatter. ReLU => v >= 0, so signed-int
    // max on the float bit pattern == float max (targets zero-initialized).
    const int n0 = col0 + ln;
    const int n1 = n0 + 16;
    {
        float bu0 = bU[f * D + n0];
        float bu1 = bU[f * D + n1];
        #pragma unroll
        for (int r = 0; r < 8; ++r) {
            int mrow = lh * 8 + r;            // D-matrix row layout
            int tgt = rowIdx[mrow];
            float v0 = accU0[r] + bu0; v0 = v0 > 0.f ? v0 : 0.f;
            float v1 = accU1[r] + bu1; v1 = v1 > 0.f ? v1 : 0.f;
            atomicMax((int*)(outAtom + (size_t)tgt * D + n0), __float_as_int(v0));
            atomicMax((int*)(outAtom + (size_t)tgt * D + n1), __float_as_int(v1));
        }
    }
    if (HAS_M) {
        float bm0 = bM[f * D + n0];
        float bm1 = bM[f * D + n1];
        #pragma unroll
        for (int r = 0; r < 8; ++r) {
            int mrow = lh * 8 + r;
            size_t base = ((size_t)f * Gdim + rowG[mrow]) * D;
            float v0 = accM0[r] + bm0; v0 = v0 > 0.f ? v0 : 0.f;
            float v1 = accM1[r] + bm1; v1 = v1 > 0.f ? v1 : 0.f;
            atomicMax((int*)(outGcl + base + n0), __float_as_int(v0));
            atomicMax((int*)(outGcl + base + n1), __float_as_int(v1));
        }
    }
}

extern "C" void kernel_launch(void* const* d_in, const int* in_sizes, int n_in,
                              void* d_out, int out_size, void* d_ws, size_t ws_size,
                              hipStream_t stream) {
    const float* x         = (const float*)d_in[0];   // [1, N, D]
    const int*   gidx      = (const int*)d_in[1];     // [F, G, 3]
    const float* first_gcs = (const float*)d_in[2];   // [F, G, D]
    const float* W_M       = (const float*)d_in[3];   // [L, F, 512, 256]
    const float* b_M       = (const float*)d_in[4];   // [L, F, 256]
    const float* W_U       = (const float*)d_in[5];   // [L, F, 512, 256]
    const float* b_U       = (const float*)d_in[6];   // [L, F, 256]
    float* out = (float*)d_out;                       // [1, N, D]

    const size_t S = (size_t)IN_DIM * D;              // 131072
    unsigned short* WT_M0 = (unsigned short*)d_ws;
    unsigned short* WT_U0 = WT_M0 + Fdim * S;
    unsigned short* WT_U1 = WT_U0 + Fdim * S;
    float* atom1 = (float*)((char*)d_ws + (size_t)3 * Fdim * S * sizeof(unsigned short));
    float* gcl1  = atom1 + (size_t)N_ATOMS * D;
    // ws usage: 3 MB weights + 51.2 MB atom1 + 204.8 MB gcl1 ~= 259 MB

    // 1) bf16 transposed weights
    int totalW = 3 * Fdim * (int)S;
    prep_weights<<<dim3((totalW + 255) / 256), dim3(256), 0, stream>>>(
        W_M, W_U, WT_M0, WT_U0, WT_U1);

    // 2) zero the scatter-max targets (must be per-call: buffers persist)
    zero4<<<dim3(2048), dim3(256), 0, stream>>>((float4*)atom1, N_ATOMS * D / 4);
    zero4<<<dim3(2048), dim3(256), 0, stream>>>((float4*)gcl1, Fdim * Gdim * (D / 4));
    zero4<<<dim3(2048), dim3(256), 0, stream>>>((float4*)out, N_ATOMS * D / 4);

    // 3) layer 0: message (clique-max) + update (atom scatter-max)
    fgnn_layer<true><<<dim3(MTILES, Fdim), dim3(256), 0, stream>>>(
        x, first_gcs, gidx, WT_M0, b_M, WT_U0, b_U, gcl1, atom1);

    // 4) layer 1: update only, into d_out
    fgnn_layer<false><<<dim3(MTILES, Fdim), dim3(256), 0, stream>>>(
        atom1, gcl1, gidx, nullptr, nullptr, WT_U1, b_U + Fdim * D, nullptr, out);
}
